// AdapterDSA_56581899157787
// MI455X (gfx1250) — matrix operand order repaired and checked by reading, not compiled, not run on hardware
//
#include <hip/hip_runtime.h>
#include <math.h>

// ---------------------------------------------------------------------------
// Problem constants (match reference)
// ---------------------------------------------------------------------------
constexpr int B_  = 2;
constexpr int C_  = 256;
constexpr int H_  = 128;
constexpr int W_  = 128;
constexpr int HW_ = H_ * W_;
constexpr int L_  = 4;
constexpr int NH_ = 8;
constexpr int NP_ = 4;
constexpr int D_  = C_ / NH_;          // 32
constexpr int M_  = B_ * HW_;          // 32768 rows for all GEMMs
constexpr int K_  = C_;                // 256 (fixed inner dim for every GEMM)

// GEMM tiling
constexpr int MT  = 128;               // block tile M
constexpr int NT  = 64;                // block tile N
constexpr int KT  = 32;                // K chunk (== WMMA K)
constexpr int NKC = K_ / KT;           // 8 K-chunks

typedef __bf16 v16bf __attribute__((ext_vector_type(16)));
typedef __bf16 v8bf  __attribute__((ext_vector_type(8)));
typedef float  v8f   __attribute__((ext_vector_type(8)));

// ---------------------------------------------------------------------------
// 2D sincos positional encoding value (channels-last index c at (h,w))
// ch = ceil(C/4)*2 = 128; first half encodes h, second half encodes w.
// ---------------------------------------------------------------------------
__device__ __forceinline__ float pe_val(int h, int w, int c) {
    int pos, cc;
    if (c < C_ / 2) { pos = h; cc = c; }
    else            { pos = w; cc = c - C_ / 2; }
    int i = cc >> 1;                                   // 0..63
    float f = __powf(10000.0f, -(float)i * (1.0f / 64.0f));
    float s = (float)pos * f;
    return (cc & 1) ? __cosf(s) : __sinf(s);
}

// ---------------------------------------------------------------------------
// key_tok[b,hw,c] = ego[b,c,h,w] + in_scale * pe(h,w,c)
// 32x32 LDS tile transpose: coalesced reads over hw, coalesced writes over c.
// grid: (HW/32, C/32, B), block: 256
// ---------------------------------------------------------------------------
__global__ __launch_bounds__(256)
void build_keytok_kernel(const float* __restrict__ ego,
                         const float* __restrict__ in_scale,
                         float* __restrict__ key_tok) {
    __shared__ float tile[32][33];                     // +1 pad: no bank conflicts
    const int b   = blockIdx.z;
    const int hw0 = blockIdx.x * 32;
    const int c0  = blockIdx.y * 32;
    const int tx  = threadIdx.x & 31;
    const int ty  = threadIdx.x >> 5;                  // 0..7

#pragma unroll
    for (int r = 0; r < 4; ++r) {
        int c = c0 + ty + r * 8;
        tile[ty + r * 8][tx] = ego[((size_t)b * C_ + c) * HW_ + hw0 + tx];
    }
    __syncthreads();

    const float is = in_scale[0];
#pragma unroll
    for (int r = 0; r < 4; ++r) {
        int hw = hw0 + ty + r * 8;
        int h  = hw / W_;
        int w  = hw - h * W_;
        int c  = c0 + tx;
        key_tok[((size_t)b * HW_ + hw) * C_ + c] =
            tile[tx][ty + r * 8] + is * pe_val(h, w, c);
    }
}

// ---------------------------------------------------------------------------
// q[b,hw,c] = out[b,hw,c] + out_scale * pe(h,w,c)   (already coalesced)
// ---------------------------------------------------------------------------
__global__ __launch_bounds__(256)
void add_pe_out_kernel(const float* __restrict__ out,
                       const float* __restrict__ out_scale,
                       float* __restrict__ q) {
    int gid = blockIdx.x * 256 + threadIdx.x;          // over B*HW*C
    int c  = gid % C_;
    int t  = gid / C_;
    int hw = t % HW_;
    int h  = hw / W_;
    int w  = hw - h * W_;
    q[gid] = out[gid] + out_scale[0] * pe_val(h, w, c);
}

// ---------------------------------------------------------------------------
// Tiled bf16 WMMA GEMM:  Out(M x N) = A(M x 256) * W(256 x N) + bias [+ res]
// - Whole transposed W panel (256 x NT bf16, 32 KB) staged in LDS ONCE.
// - A tiles double-buffered; next chunk's global loads issued before WMMAs.
// Block: 256 threads = 8 waves; block tile 128x64; wave tile 32x32 (2x2 frags).
// ---------------------------------------------------------------------------
__global__ __launch_bounds__(256)
void gemm_bf16_wmma(const float* __restrict__ A, const float* __restrict__ Wm,
                    const float* __restrict__ bias, const float* __restrict__ res,
                    float* __restrict__ Out, int N) {
    __shared__ alignas(16) __bf16 Wp[NT * K_];      // transposed [n][k], 32 KB
    __shared__ alignas(16) __bf16 As[2][MT * KT];   // double-buffered, 2 x 8 KB

    const int tid     = threadIdx.x;
    const int bm      = blockIdx.x * MT;
    const int bn      = blockIdx.y * NT;
    const int wave    = tid >> 5;
    const int lane    = tid & 31;
    const int laneLow = lane & 15;
    const int laneHi  = lane >> 4;                  // 0 or 1
    const int wm      = (wave >> 1) * 32;           // 0,32,64,96
    const int wn      = (wave & 1) * 32;            // 0,32

    // ---- stage full W panel transposed, fp32 -> bf16 (once) ----
#pragma unroll
    for (int it = 0; it < 16; ++it) {
        int linear = it * 256 + tid;                // 0..4095
        int k  = linear >> 4;                       // 0..255
        int ng = (linear & 15) * 4;                 // 0,4,...,60
        int n  = bn + ng;
        if (n + 3 < N) {
            float4 f = *(const float4*)(Wm + (size_t)k * N + n);
            Wp[(ng + 0) * K_ + k] = (__bf16)f.x;
            Wp[(ng + 1) * K_ + k] = (__bf16)f.y;
            Wp[(ng + 2) * K_ + k] = (__bf16)f.z;
            Wp[(ng + 3) * K_ + k] = (__bf16)f.w;
        } else {
#pragma unroll
            for (int j = 0; j < 4; ++j) {
                float v = (n + j < N) ? Wm[(size_t)k * N + n + j] : 0.0f;
                Wp[(ng + j) * K_ + k] = (__bf16)v;
            }
        }
    }

    // A staging helpers: thread owns row m = tid/2, 16-float half kk
    const int sm = tid >> 1;
    const int sk = (tid & 1) * 16;
    const float* arow = A + (size_t)(bm + sm) * K_ + sk;

    float4 st[4];
    auto load_stage = [&](int kc) {
        const float* src = arow + kc * KT;
        st[0] = *(const float4*)(src + 0);
        st[1] = *(const float4*)(src + 4);
        st[2] = *(const float4*)(src + 8);
        st[3] = *(const float4*)(src + 12);
        if (kc + 2 < NKC)                            // prefetch 2 chunks ahead
            __builtin_prefetch(src + 2 * KT, 0, 0);
    };
    auto store_stage = [&](int buf) {
        __bf16* dst = &As[buf][sm * KT + sk];
#pragma unroll
        for (int j = 0; j < 4; ++j) {
            dst[4 * j + 0] = (__bf16)st[j].x;  dst[4 * j + 1] = (__bf16)st[j].y;
            dst[4 * j + 2] = (__bf16)st[j].z;  dst[4 * j + 3] = (__bf16)st[j].w;
        }
    };

    v8f acc[2][2] = {};

    load_stage(0);
    store_stage(0);
    __syncthreads();

    for (int kc = 0; kc < NKC; ++kc) {
        const int cur = kc & 1;
        if (kc + 1 < NKC) load_stage(kc + 1);       // global loads in flight

        // ---- fragment loads (documented 16-bit WMMA layouts) ----
        // A 16x32: lanes 0-15 row M=lane, elems K {0..7,16..23}; lanes 16-31 +8
        v16bf afrag[2], bfrag[2];
#pragma unroll
        for (int fm = 0; fm < 2; ++fm) {
            int m = wm + fm * 16 + laneLow;
            v8bf lo = *(const v8bf*)&As[cur][m * KT + laneHi * 8];
            v8bf hi = *(const v8bf*)&As[cur][m * KT + 16 + laneHi * 8];
#pragma unroll
            for (int e = 0; e < 8; ++e) { afrag[fm][e] = lo[e]; afrag[fm][e + 8] = hi[e]; }
        }
        // B 32x16: lanes 0-15 col N=lane, elems K {0..7,16..23}; lanes 16-31 +8
#pragma unroll
        for (int fn = 0; fn < 2; ++fn) {
            int n = wn + fn * 16 + laneLow;
            v8bf lo = *(const v8bf*)&Wp[n * K_ + kc * KT + laneHi * 8];
            v8bf hi = *(const v8bf*)&Wp[n * K_ + kc * KT + 16 + laneHi * 8];
#pragma unroll
            for (int e = 0; e < 8; ++e) { bfrag[fn][e] = lo[e]; bfrag[fn][e + 8] = hi[e]; }
        }

#pragma unroll
        for (int fm = 0; fm < 2; ++fm)
#pragma unroll
            for (int fn = 0; fn < 2; ++fn)
                acc[fm][fn] = __builtin_amdgcn_wmma_f32_16x16x32_bf16(
                    false, afrag[fm], false, bfrag[fn],
                    (short)0, acc[fm][fn], false, false);

        if (kc + 1 < NKC) store_stage(cur ^ 1);     // fill the other buffer
        __syncthreads();
    }

    // ---- epilogue: bias (+ residual), fp32 store ----
#pragma unroll
    for (int fm = 0; fm < 2; ++fm) {
#pragma unroll
        for (int fn = 0; fn < 2; ++fn) {
            int n = bn + wn + fn * 16 + laneLow;
            if (n < N) {
                float bv = bias ? bias[n] : 0.0f;
#pragma unroll
                for (int r = 0; r < 8; ++r) {
                    int m = bm + wm + fm * 16 + laneHi * 8 + r;
                    size_t o = (size_t)m * N + n;
                    float val = acc[fm][fn][r] + bv;
                    if (res) val += res[o];
                    Out[o] = val;
                }
            }
        }
    }
}

// ---------------------------------------------------------------------------
// Deformable attention: one thread per (b, query, head).
// softmax over NP logits, bilinear-gather from v, accumulate D=32 channels.
// ---------------------------------------------------------------------------
__global__ __launch_bounds__(256)
void deform_attn_kernel(const float* __restrict__ V,      // (B*HW, C) = (B,HW,NH,D)
                        const float* __restrict__ offs,   // (B*HW, NH*NP*2)
                        const float* __restrict__ awl,    // (B*HW, NH*NP) logits
                        float* __restrict__ attn) {       // (B*HW, C)
    int idx = blockIdx.x * 256 + threadIdx.x;             // (b*HW+hw)*NH + nh
    int nh = idx & (NH_ - 1);
    int t  = idx >> 3;                                    // b*HW + hw
    int b  = t / HW_;
    int hw = t - b * HW_;
    int py = hw / W_;
    int px = hw - py * W_;

    // softmax over NP=4
    const float* lg = awl + (size_t)t * (NH_ * NP_) + nh * NP_;
    float l0 = lg[0], l1 = lg[1], l2 = lg[2], l3 = lg[3];
    float mx = fmaxf(fmaxf(l0, l1), fmaxf(l2, l3));
    float e0 = __expf(l0 - mx), e1 = __expf(l1 - mx);
    float e2 = __expf(l2 - mx), e3 = __expf(l3 - mx);
    float inv = 1.0f / (e0 + e1 + e2 + e3);
    float wp[NP_] = { e0 * inv, e1 * inv, e2 * inv, e3 * inv };

    const float* of = offs + (size_t)t * (NH_ * NP_ * 2) + nh * (NP_ * 2);
    float refx = (0.5f + px) * (1.0f / W_);
    float refy = (0.5f + py) * (1.0f / H_);

    float4 acc4[D_ / 4] = {};
    const float* vb = V + (size_t)b * HW_ * C_ + nh * D_;

#pragma unroll
    for (int p = 0; p < NP_; ++p) {
        float lx = refx + of[p * 2 + 0] * (1.0f / W_);
        float ly = refy + of[p * 2 + 1] * (1.0f / H_);
        float ix = lx * W_ - 0.5f;
        float iy = ly * H_ - 0.5f;
        float x0 = floorf(ix), y0 = floorf(iy);
        float fx = ix - x0,    fy = iy - y0;
        float cw[4] = { (1.0f - fx) * (1.0f - fy) * wp[p],
                        fx * (1.0f - fy)          * wp[p],
                        (1.0f - fx) * fy          * wp[p],
                        fx * fy                   * wp[p] };
        float cx[4] = { x0, x0 + 1.0f, x0,        x0 + 1.0f };
        float cy[4] = { y0, y0,        y0 + 1.0f, y0 + 1.0f };
#pragma unroll
        for (int c2 = 0; c2 < 4; ++c2) {
            bool valid = (cx[c2] >= 0.0f) && (cx[c2] <= (float)(W_ - 1)) &&
                         (cy[c2] >= 0.0f) && (cy[c2] <= (float)(H_ - 1));
            if (valid) {
                int xi = (int)cx[c2];
                int yi = (int)cy[c2];
                const float* row = vb + (size_t)(yi * W_ + xi) * C_;
                float w = cw[c2];
#pragma unroll
                for (int dd = 0; dd < D_ / 4; ++dd) {
                    float4 vv = *(const float4*)(row + dd * 4);
                    acc4[dd].x = fmaf(w, vv.x, acc4[dd].x);
                    acc4[dd].y = fmaf(w, vv.y, acc4[dd].y);
                    acc4[dd].z = fmaf(w, vv.z, acc4[dd].z);
                    acc4[dd].w = fmaf(w, vv.w, acc4[dd].w);
                }
            }
        }
    }

    float* o = attn + (size_t)t * C_ + nh * D_;
#pragma unroll
    for (int dd = 0; dd < D_ / 4; ++dd)
        *(float4*)(o + dd * 4) = acc4[dd];
}

// ---------------------------------------------------------------------------
// Orchestration
// ---------------------------------------------------------------------------
extern "C" void kernel_launch(void* const* d_in, const int* in_sizes, int n_in,
                              void* d_out, int out_size, void* d_ws, size_t ws_size,
                              hipStream_t stream) {
    const float* ego    = (const float*)d_in[0];
    const float* conv_w = (const float*)d_in[1];
    const float* conv_b = (const float*)d_in[2];
    const float* in_s   = (const float*)d_in[3];
    const float* out_s  = (const float*)d_in[4];
    const float* off_w  = (const float*)d_in[5];
    const float* off_b  = (const float*)d_in[6];
    const float* aw_w   = (const float*)d_in[7];
    const float* aw_b   = (const float*)d_in[8];
    const float* vp_w   = (const float*)d_in[9];
    const float* vp_b   = (const float*)d_in[10];
    const float* op_w   = (const float*)d_in[11];
    const float* op_b   = (const float*)d_in[12];

    // workspace carve-out (fp32)
    float* ws = (float*)d_ws;
    const size_t MC = (size_t)M_ * C_;
    float* key_tok = ws; ws += MC;               // (M, C)
    float* outb    = ws; ws += MC;               // (M, C)
    float* qb      = ws; ws += MC;               // (M, C)
    float* vbuf    = ws; ws += MC;               // (M, C)
    float* attn    = ws; ws += MC;               // (M, C)
    float* offsb   = ws; ws += (size_t)M_ * (NH_ * NP_ * 2);  // (M, 64)
    float* awb     = ws; ws += (size_t)M_ * (NH_ * NP_);      // (M, 32)

    const dim3 blk(256);
    const int ew_blocks = (B_ * HW_ * C_) / 256;                // 32768
    const dim3 gTok(HW_ / 32, C_ / 32, B_);                     // (512, 8, 2)
    const dim3 gFull(M_ / MT, (C_ + NT - 1) / NT);              // (256, 4)
    const dim3 gOff (M_ / MT, (NH_ * NP_ * 2 + NT - 1) / NT);   // (256, 1)
    const dim3 gAw  (M_ / MT, (NH_ * NP_ + NT - 1) / NT);       // (256, 1)

    // key tokens with input positional encoding (coalesced transpose)
    build_keytok_kernel<<<gTok, blk, 0, stream>>>(ego, in_s, key_tok);

    // 1x1 conv == GEMM over tokens
    gemm_bf16_wmma<<<gFull, blk, 0, stream>>>(key_tok, conv_w, conv_b, nullptr,
                                              outb, C_);

    for (int l = 0; l < L_; ++l) {
        // q = out + out_scale * PE
        add_pe_out_kernel<<<ew_blocks, blk, 0, stream>>>(outb, out_s, qb);

        // value projection (from key tokens)
        gemm_bf16_wmma<<<gFull, blk, 0, stream>>>(
            key_tok, vp_w + (size_t)l * C_ * C_, vp_b + l * C_, nullptr,
            vbuf, C_);

        // sampling offsets: N = 64
        gemm_bf16_wmma<<<gOff, blk, 0, stream>>>(
            qb, off_w + (size_t)l * C_ * (NH_ * NP_ * 2), off_b + l * (NH_ * NP_ * 2),
            nullptr, offsb, NH_ * NP_ * 2);

        // attention-weight logits: N = 32 (softmax fused into deform kernel)
        gemm_bf16_wmma<<<gAw, blk, 0, stream>>>(
            qb, aw_w + (size_t)l * C_ * (NH_ * NP_), aw_b + l * (NH_ * NP_),
            nullptr, awb, NH_ * NP_);

        // deformable bilinear sampling + weighted sum
        deform_attn_kernel<<<(B_ * HW_ * NH_) / 256, blk, 0, stream>>>(
            vbuf, offsb, awb, attn);

        // output projection + residual (identity = q); last layer -> d_out
        float* dst = (l == L_ - 1) ? (float*)d_out : outb;
        gemm_bf16_wmma<<<gFull, blk, 0, stream>>>(
            attn, op_w + (size_t)l * C_ * C_, op_b + l * C_, qb,
            dst, C_);
    }
}
